// LoRA_MoElayer_89094801588664
// MI455X (gfx1250) — matrix operand order repaired and checked by reading, not compiled, not run on hardware
//
#include <hip/hip_runtime.h>

// ---------------- problem constants ----------------
#define BN    8192      // B*N tokens
#define CDIM  768       // C
#define OUTC  2304      // 3*C
#define NEXP  7
#define NPAD  416       // padded sum of lora dims (392 -> 400 -> 416 so K%32==0)

typedef __attribute__((ext_vector_type(16))) __bf16 bf16x16;
typedef __attribute__((ext_vector_type(8)))  __bf16 bf16x8;
typedef __attribute__((ext_vector_type(8)))  float  floatx8;

// padded expert column layout: offsets multiples of 16
//  e0:[0,8) (pad 8..15)  e1:[16,32) e2:[32,64) e3:[64,112) e4:[112,176)
//  e5:[176,272) e6:[272,400)  pad:[400,416)
__device__ __forceinline__ int expert_of(int n, int& j) {
    if (n <   8) { j = n;       return 0; }
    if (n <  16) {              return -1; }
    if (n <  32) { j = n -  16; return 1; }
    if (n <  64) { j = n -  32; return 2; }
    if (n < 112) { j = n -  64; return 3; }
    if (n < 176) { j = n - 112; return 4; }
    if (n < 272) { j = n - 176; return 5; }
    if (n < 400) { j = n - 272; return 6; }
    return -1;
}

// branchless 16-col-tile -> expert map (-1 = zero padding tile)
__constant__ int c_tile2e[26] = {0, 1, 2, 2, 3, 3, 3, 4, 4, 4, 4,
                                 5, 5, 5, 5, 5, 5, 6, 6, 6, 6, 6, 6, 6, 6, -1};

// ---------------- gating precompute ----------------
// EK[e,c2] = (expert_emb @ Wk)[e,c2]
__global__ __launch_bounds__(256) void ek_kernel(const float* __restrict__ emb,
                                                 const float* __restrict__ Wk,
                                                 float* __restrict__ EK) {
    int t = blockIdx.x * 256 + threadIdx.x;
    if (t >= NEXP * CDIM) return;
    int e = t / CDIM, c2 = t % CDIM;
    float s = 0.f;
    for (int c3 = 0; c3 < CDIM; ++c3)
        s += emb[e * CDIM + c3] * Wk[c3 * CDIM + c2];   // coalesced in c2
    EK[e * CDIM + c2] = s;
}

// G[c,e] = (scale/4) * sum_c2 Wq[c,c2] * EK[e,c2]   (scale = (C/4)^-0.5 = 192^-0.5)
__global__ __launch_bounds__(256) void g_kernel(const float* __restrict__ Wq,
                                                const float* __restrict__ EK,
                                                float* __restrict__ G) {
    int t = blockIdx.x * 256 + threadIdx.x;
    if (t >= CDIM * NEXP) return;
    int c = t / NEXP, e = t % NEXP;
    float s = 0.f;
    for (int c2 = 0; c2 < CDIM; ++c2)
        s += Wq[c * CDIM + c2] * EK[e * CDIM + c2];
    G[t] = s * (0.25f * 0.0721687836487032f);           // 0.25/sqrt(192)
}

// gates[row, :] = softmax(x[row,:] @ G)   one wave32 per row
__global__ __launch_bounds__(256) void gates_kernel(const float* __restrict__ x,
                                                    const float* __restrict__ G,
                                                    float* __restrict__ gates) {
    const int wave = threadIdx.x >> 5;
    const int lane = threadIdx.x & 31;
    const int row  = blockIdx.x * 8 + wave;
    const float* xr = x + (size_t)row * CDIM;
    float p[NEXP] = {0.f, 0.f, 0.f, 0.f, 0.f, 0.f, 0.f};
    for (int c = lane; c < CDIM; c += 32) {
        float xv = xr[c];
        const float* g = G + c * NEXP;
#pragma unroll
        for (int e = 0; e < NEXP; ++e) p[e] += xv * g[e];
    }
#pragma unroll
    for (int e = 0; e < NEXP; ++e)
        for (int off = 16; off > 0; off >>= 1)
            p[e] += __shfl_xor(p[e], off, 32);
    if (lane == 0) {
        float mx = p[0];
#pragma unroll
        for (int e = 1; e < NEXP; ++e) mx = fmaxf(mx, p[e]);
        float ex[NEXP], s = 0.f;
#pragma unroll
        for (int e = 0; e < NEXP; ++e) { ex[e] = __expf(p[e] - mx); s += ex[e]; }
        float inv = 1.f / s;
#pragma unroll
        for (int e = 0; e < NEXP; ++e) gates[row * NEXP + e] = ex[e] * inv;
    }
}

// ---------------- weight packing (fp32 -> bf16, transposed, zero-padded) ----------------
// AcatT[n, c] = A_e[c, j]  (n -> expert e, local col j), shape (NPAD x CDIM)
__global__ __launch_bounds__(256) void packA_kernel(const float* __restrict__ A0, const float* __restrict__ A1,
                                                    const float* __restrict__ A2, const float* __restrict__ A3,
                                                    const float* __restrict__ A4, const float* __restrict__ A5,
                                                    const float* __restrict__ A6, __bf16* __restrict__ AcatT) {
    int t = blockIdx.x * 256 + threadIdx.x;
    if (t >= NPAD * CDIM) return;
    int n = t / CDIM, c = t % CDIM;
    const float* As[NEXP] = {A0, A1, A2, A3, A4, A5, A6};
    const int dims[NEXP]  = {8, 16, 32, 48, 64, 96, 128};
    int j; int e = expert_of(n, j);
    float v = (e >= 0) ? As[e][c * dims[e] + j] : 0.f;
    AcatT[t] = (__bf16)v;
}

// BcatT[m, kg] = B_e[j, m]  (kg -> expert e, local row j), shape (OUTC x NPAD)
__global__ __launch_bounds__(256) void packB_kernel(const float* __restrict__ B0, const float* __restrict__ B1,
                                                    const float* __restrict__ B2, const float* __restrict__ B3,
                                                    const float* __restrict__ B4, const float* __restrict__ B5,
                                                    const float* __restrict__ B6, __bf16* __restrict__ BcatT) {
    int t = blockIdx.x * 256 + threadIdx.x;
    if (t >= OUTC * NPAD) return;
    int m = t % OUTC, kg = t / OUTC;     // consecutive threads -> coalesced B_e reads
    const float* Bs[NEXP] = {B0, B1, B2, B3, B4, B5, B6};
    int j; int e = expert_of(kg, j);
    float v = (e >= 0) ? Bs[e][j * OUTC + m] : 0.f;
    BcatT[m * NPAD + kg] = (__bf16)v;
}

// ---------------- GEMM1: U = (x @ Acat) * gate_block, bf16 out ----------------
// one wave per 16x32 strip (2 col tiles share the converted A fragment); K = 768
__global__ __launch_bounds__(256) void gemm1_kernel(const float* __restrict__ x,
                                                    const __bf16* __restrict__ AcatT,
                                                    const float* __restrict__ gates,
                                                    __bf16* __restrict__ U) {
    const int wave = threadIdx.x >> 5;
    const int lane = threadIdx.x & 31;
    const int half = lane >> 4;
    const int l16  = lane & 15;
    const int id  = blockIdx.x * 8 + wave;           // 512*13 strips, grid*8 exact
    const int rt  = id / 13, cp = id % 13;           // cp = pair of 16-col tiles
    const int row0 = rt * 16;
    const int n0   = cp * 32;

    floatx8 acc0 = {0.f, 0.f, 0.f, 0.f, 0.f, 0.f, 0.f, 0.f};
    floatx8 acc1 = acc0;
    const float*  xp  = x + (size_t)(row0 + l16) * CDIM + half * 16;         // A-frag row=l16
    const __bf16* bp0 = AcatT + (size_t)(n0 +  0 + l16) * CDIM + half * 16;  // B-frag col=l16
    const __bf16* bp1 = AcatT + (size_t)(n0 + 16 + l16) * CDIM + half * 16;

#pragma unroll 2
    for (int k0 = 0; k0 < CDIM; k0 += 32) {
        // A fragment: 16 bf16 per lane; elem j -> K=k0+half*16+j
        float4 a0 = *(const float4*)(xp + k0);
        float4 a1 = *(const float4*)(xp + k0 + 4);
        float4 a2 = *(const float4*)(xp + k0 + 8);
        float4 a3 = *(const float4*)(xp + k0 + 12);
        bf16x16 a;
        a[0] = (__bf16)a0.x;  a[1] = (__bf16)a0.y;  a[2]  = (__bf16)a0.z;  a[3]  = (__bf16)a0.w;
        a[4] = (__bf16)a1.x;  a[5] = (__bf16)a1.y;  a[6]  = (__bf16)a1.z;  a[7]  = (__bf16)a1.w;
        a[8] = (__bf16)a2.x;  a[9] = (__bf16)a2.y;  a[10] = (__bf16)a2.z;  a[11] = (__bf16)a2.w;
        a[12] = (__bf16)a3.x; a[13] = (__bf16)a3.y; a[14] = (__bf16)a3.z;  a[15] = (__bf16)a3.w;
        // B fragments: 16 contiguous K (halves split at K=16) -> aligned 32B loads
        bf16x16 b0 = *(const bf16x16*)(bp0 + k0);
        acc0 = __builtin_amdgcn_wmma_f32_16x16x32_bf16(false, a, false, b0, (short)0, acc0,
                                                       false, false);
        bf16x16 b1 = *(const bf16x16*)(bp1 + k0);
        acc1 = __builtin_amdgcn_wmma_f32_16x16x32_bf16(false, a, false, b1, (short)0, acc1,
                                                       false, false);
    }

    // branchless gate scale: each 16-col tile belongs to one expert (padded offsets)
    const int e0 = c_tile2e[cp * 2];
    const int e1 = c_tile2e[cp * 2 + 1];
    const int e0c = (e0 < 0) ? 0 : e0;
    const int e1c = (e1 < 0) ? 0 : e1;
#pragma unroll
    for (int r = 0; r < 8; ++r) {
        int m = row0 + r + half * 8;                 // C-layout: VGPR r, half selects M+8
        float g0 = gates[m * NEXP + e0c]; if (e0 < 0) g0 = 0.f;
        float g1 = gates[m * NEXP + e1c]; if (e1 < 0) g1 = 0.f;
        U[(size_t)m * NPAD + n0 +  0 + l16] = (__bf16)(acc0[r] * g0);
        U[(size_t)m * NPAD + n0 + 16 + l16] = (__bf16)(acc1[r] * g1);
    }
}

// ---------------- GEMM2: Y = U @ Bcat ----------------
// one wave per 16x64 strip (4 accumulators share the A fragment); K = 416, 13 steps
__global__ __launch_bounds__(256) void gemm2_kernel(const __bf16* __restrict__ U,
                                                    const __bf16* __restrict__ BcatT,
                                                    float* __restrict__ Y) {
    const int wave = threadIdx.x >> 5;
    const int lane = threadIdx.x & 31;
    const int half = lane >> 4;
    const int l16  = lane & 15;
    const int id = blockIdx.x * 8 + wave;            // 512*36 waves, grid*8 exact
    const int rt = id / 36, cg = id % 36;
    const int row0 = rt * 16;
    const int col0 = cg * 64;

    floatx8 acc0 = {0.f,0.f,0.f,0.f,0.f,0.f,0.f,0.f};
    floatx8 acc1 = acc0, acc2 = acc0, acc3 = acc0;

    const __bf16* ap  = U + (size_t)(row0 + l16) * NPAD + half * 16;
    const __bf16* bp0 = BcatT + (size_t)(col0 +  0 + l16) * NPAD + half * 16;
    const __bf16* bp1 = BcatT + (size_t)(col0 + 16 + l16) * NPAD + half * 16;
    const __bf16* bp2 = BcatT + (size_t)(col0 + 32 + l16) * NPAD + half * 16;
    const __bf16* bp3 = BcatT + (size_t)(col0 + 48 + l16) * NPAD + half * 16;

    for (int k0 = 0; k0 < NPAD; k0 += 32) {
        bf16x8 lo = *(const bf16x8*)(ap + k0);        // K = k0 + half*16 + j
        bf16x8 hi = *(const bf16x8*)(ap + k0 + 8);    // K = k0 + half*16 + 8 + j
        bf16x16 a = __builtin_shufflevector(lo, hi, 0, 1, 2, 3, 4, 5, 6, 7,
                                            8, 9, 10, 11, 12, 13, 14, 15);
        bf16x16 b0 = *(const bf16x16*)(bp0 + k0);
        acc0 = __builtin_amdgcn_wmma_f32_16x16x32_bf16(false, a, false, b0, (short)0, acc0, false, false);
        bf16x16 b1 = *(const bf16x16*)(bp1 + k0);
        acc1 = __builtin_amdgcn_wmma_f32_16x16x32_bf16(false, a, false, b1, (short)0, acc1, false, false);
        bf16x16 b2 = *(const bf16x16*)(bp2 + k0);
        acc2 = __builtin_amdgcn_wmma_f32_16x16x32_bf16(false, a, false, b2, (short)0, acc2, false, false);
        bf16x16 b3 = *(const bf16x16*)(bp3 + k0);
        acc3 = __builtin_amdgcn_wmma_f32_16x16x32_bf16(false, a, false, b3, (short)0, acc3, false, false);
    }

    // Y is a 75MB write-once stream: keep it out of the cache hierarchy
#pragma unroll
    for (int r = 0; r < 8; ++r) {
        int m = row0 + r + half * 8;
        float* yr = Y + (size_t)m * OUTC + col0 + l16;
        __builtin_nontemporal_store(acc0[r], yr);
        __builtin_nontemporal_store(acc1[r], yr + 16);
        __builtin_nontemporal_store(acc2[r], yr + 32);
        __builtin_nontemporal_store(acc3[r], yr + 48);
    }
}

// ---------------- importance / CV^2 loss ----------------
__global__ __launch_bounds__(256) void loss_kernel(const float* __restrict__ gates,
                                                   float* __restrict__ out_loss) {
    __shared__ float sh[NEXP * 256];
    const int tid = threadIdx.x;
    float p[NEXP] = {0.f, 0.f, 0.f, 0.f, 0.f, 0.f, 0.f};
    for (int r = tid; r < BN; r += 256) {
#pragma unroll
        for (int e = 0; e < NEXP; ++e) p[e] += gates[r * NEXP + e];
    }
#pragma unroll
    for (int e = 0; e < NEXP; ++e) sh[e * 256 + tid] = p[e];
    __syncthreads();
    for (int s = 128; s > 0; s >>= 1) {
        if (tid < s) {
#pragma unroll
            for (int e = 0; e < NEXP; ++e) sh[e * 256 + tid] += sh[e * 256 + tid + s];
        }
        __syncthreads();
    }
    if (tid == 0) {
        float imp[NEXP], mu = 0.f;
#pragma unroll
        for (int e = 0; e < NEXP; ++e) { imp[e] = sh[e * 256]; mu += imp[e]; }
        mu *= (1.f / NEXP);
        float var = 0.f;
#pragma unroll
        for (int e = 0; e < NEXP; ++e) { float d = imp[e] - mu; var += d * d; }
        var *= (1.f / (NEXP - 1));                    // ddof=1
        float cv2 = var / (mu * mu + 1e-10f);
        out_loss[0] = 2.f * cv2;                      // cv2 + cv2
    }
}

// ---------------- launcher ----------------
extern "C" void kernel_launch(void* const* d_in, const int* in_sizes, int n_in,
                              void* d_out, int out_size, void* d_ws, size_t ws_size,
                              hipStream_t stream) {
    const float* x   = (const float*)d_in[0];
    const float* Wq  = (const float*)d_in[1];
    const float* Wk  = (const float*)d_in[2];
    const float* emb = (const float*)d_in[3];
    // setup_inputs order interleaves A_i / B_i
    const float* A[NEXP]; const float* B[NEXP];
    for (int i = 0; i < NEXP; ++i) {
        A[i] = (const float*)d_in[4 + 2 * i];
        B[i] = (const float*)d_in[5 + 2 * i];
    }

    char* ws = (char*)d_ws;                            // all offsets 256B-aligned
    float*  EK    = (float*)(ws + 0);                  //  7*768*4   = 21504
    float*  G     = (float*)(ws + 21504);              //  768*7*4   = 21504
    float*  gates = (float*)(ws + 43008);              //  8192*7*4  = 229376
    __bf16* AcatT = (__bf16*)(ws + 272384);            //  416*768*2 = 638976
    __bf16* BcatT = (__bf16*)(ws + 911360);            //  2304*416*2= 1916928
    __bf16* U     = (__bf16*)(ws + 2828288);           //  8192*416*2= 6815744 (end ~9.6MB)

    float* Y    = (float*)d_out;
    float* loss = Y + (size_t)BN * OUTC;

    ek_kernel   <<<21,   256, 0, stream>>>(emb, Wk, EK);
    g_kernel    <<<21,   256, 0, stream>>>(Wq, EK, G);
    gates_kernel<<<1024, 256, 0, stream>>>(x, G, gates);
    packA_kernel<<<1248, 256, 0, stream>>>(A[0], A[1], A[2], A[3], A[4], A[5], A[6], AcatT);
    packB_kernel<<<3744, 256, 0, stream>>>(B[0], B[1], B[2], B[3], B[4], B[5], B[6], BcatT);
    gemm1_kernel<<<832,  256, 0, stream>>>(x, AcatT, gates, U);   // 512*13 strips / 8 waves
    gemm2_kernel<<<2304, 256, 0, stream>>>(U, BcatT, Y);          // 512*36 strips / 8 waves
    loss_kernel <<<1,    256, 0, stream>>>(gates, loss);
}
